// VQVAE_78812649882290
// MI455X (gfx1250) — matrix operand order repaired and checked by reading, not compiled, not run on hardware
//
#include <hip/hip_runtime.h>
#include <hip/hip_bf16.h>

// ---------------------------------------------------------------------------
// CDNA5 (gfx1250) VQ-VAE forward.  All convolutions run as implicit GEMM on
// the wave32 WMMA bf16 pipe (v_wmma_f32_16x16x32_bf16, fp32 accumulate).
// Inter-layer activations are stored bf16 (memory-bound at 23.3 TB/s ->
// halve HBM bytes); fp32 kept for the VQ input z and the final output.
// Each wave owns a 16(Cout) x 64(pixel) tile = 4 accumulators, so one
// A-fragment feeds 4 WMMAs per K-chunk.  Conv geometry is templated so all
// im2col address math lowers to shifts/masks (spatial dims are powers of 2).
// ---------------------------------------------------------------------------

typedef __attribute__((ext_vector_type(8)))  float   v8f;
typedef __attribute__((ext_vector_type(8)))  __bf16  v8bf;
typedef __attribute__((ext_vector_type(16))) __bf16  v16bf;

#define TILE_P 512          // output pixels per workgroup (8 waves x 64)
#define TILE_K 32           // K chunk per WMMA
#define BROW   48           // padded LDS row stride (bf16): 96B keeps 16B align, spreads banks

// TI/TO: activation storage types (float or __bf16).
// KH,KW,STRIDE,PAD: conv geometry.  LDIL==2: lhs-dilated conv (ConvTranspose2d k4 s2 p1).
// WOB = log2(Wo), HWOB = log2(Ho*Wo).
template<typename TI, typename TO, int KH, int KW, int STRIDE, int PAD, int LDIL, int WOB, int HWOB>
__global__ __launch_bounds__(256) void conv_wmma_bf16(
    const TI* __restrict__ X, const float* __restrict__ W,
    const float* __restrict__ Bv, const TO* __restrict__ R,
    TO* __restrict__ Y,
    int N, int Cin, int H, int Wi, int Cout,
    int relu_in, int relu_out)
{
    constexpr int KHKW = KH * KW;
    constexpr int Wo   = 1 << WOB;
    constexpr int HWo  = 1 << HWOB;

    __shared__ __align__(16) __bf16 Ws[16 * TILE_K];     // weight tile  [16 co][32 k]
    __shared__ __align__(16) __bf16 Bs[TILE_P * BROW];   // im2col tile, pixel-major [512 p][32 k]

    const int tid  = threadIdx.x;
    const int wave = tid >> 5;
    const int lane = tid & 31;
    const int hgrp = lane >> 4;      // lane group 0/1
    const int mcol = lane & 15;      // A row / B column within a 16x16 tile

    const int P    = N << HWOB;
    const int Ktot = Cin * KHKW;
    const int pixBase = blockIdx.x * TILE_P;
    const int coBase  = blockIdx.y * 16;

    // Per-thread staged pixels (tid and tid+256), decomposition hoisted out of K loop.
    int  s_n[2], s_ho[2], s_wo[2];
    bool s_v[2];
    #pragma unroll
    for (int pi = 0; pi < 2; ++pi) {
        int pg = pixBase + tid + pi * 256;
        s_v[pi] = pg < P;
        int pc = s_v[pi] ? pg : 0;
        s_n[pi]  = pc >> HWOB;
        int shw  = pc & (HWo - 1);
        s_ho[pi] = shw >> WOB;
        s_wo[pi] = shw & (Wo - 1);
    }

    v8f acc[4] = {};

    for (int k0 = 0; k0 < Ktot; k0 += TILE_K) {
        // ---- stage weight tile (512 bf16, 2 per thread), zero-padded -------
        #pragma unroll
        for (int j = 0; j < 2; ++j) {
            int idx = tid + j * 256;
            int m  = idx >> 5;
            int kk = idx & 31;
            int kg = k0 + kk;
            int co = coBase + m;
            float wv = 0.f;
            if (co < Cout && kg < Ktot) wv = W[co * Ktot + kg];
            Ws[m * TILE_K + kk] = (__bf16)wv;
        }
        // ---- stage im2col tile: this thread fills 2 full pixel columns -----
        #pragma unroll
        for (int pi = 0; pi < 2; ++pi) {
            const int prow = tid + pi * 256;
            __bf16* dst = &Bs[prow * BROW];
            for (int kk = 0; kk < TILE_K; ++kk) {
                int kg = k0 + kk;
                float v = 0.f;
                if (s_v[pi] && kg < Ktot) {
                    int ci = kg / KHKW;            // compile-time divisor
                    int rs = kg - ci * KHKW;
                    int r  = rs / KW;
                    int s  = rs - r * KW;
                    int hn = s_ho[pi] * STRIDE - PAD + r;
                    int wn = s_wo[pi] * STRIDE - PAD + s;
                    if constexpr (LDIL == 2) {
                        if (hn >= 0 && wn >= 0 && !(hn & 1) && !(wn & 1)) {
                            int hi = hn >> 1, wi = wn >> 1;
                            if (hi < H && wi < Wi)
                                v = (float)X[((s_n[pi] * Cin + ci) * H + hi) * Wi + wi];
                        }
                    } else {
                        if (hn >= 0 && wn >= 0 && hn < H && wn < Wi)
                            v = (float)X[((s_n[pi] * Cin + ci) * H + hn) * Wi + wn];
                    }
                    if (relu_in) v = v > 0.f ? v : 0.f;
                }
                dst[kk] = (__bf16)v;
            }
        }
        __syncthreads();

        // ---- fragments per ISA 7.12.2 layouts (two b128 LDS loads each) ----
        union { v16bf v; v8bf h[2]; } afrag;
        const __bf16* wrow = &Ws[mcol * TILE_K];
        afrag.h[0] = *(const v8bf*)(wrow + 8 * hgrp);        // K = 8*half + 0..7
        afrag.h[1] = *(const v8bf*)(wrow + 16 + 8 * hgrp);   // K = 16 + 8*half + 0..7

        #pragma unroll
        for (int a = 0; a < 4; ++a) {
            union { v16bf v; v8bf h[2]; } bfrag;
            const __bf16* brow = &Bs[(wave * 64 + a * 16 + mcol) * BROW];
            bfrag.h[0] = *(const v8bf*)(brow + 8 * hgrp);        // K = 8*half + 0..7
            bfrag.h[1] = *(const v8bf*)(brow + 16 + 8 * hgrp);   // K = 16 + 8*half + 0..7
            acc[a] = __builtin_amdgcn_wmma_f32_16x16x32_bf16(
                         false, afrag.v, false, bfrag.v, (short)0, acc[a], false, false);
        }
        __syncthreads();
    }

    // ---- epilogue: bias + optional residual + optional ReLU ----------------
    #pragma unroll
    for (int a = 0; a < 4; ++a) {
        const int opix = pixBase + wave * 64 + a * 16 + mcol;
        const bool ov  = opix < P;
        const int op   = ov ? opix : 0;
        const int on   = op >> HWOB;
        const int ohw  = op & (HWo - 1);
        #pragma unroll
        for (int v = 0; v < 8; ++v) {
            int co = coBase + v + 8 * hgrp;
            if (ov && co < Cout) {
                int oidx = ((on * Cout + co) << HWOB) + ohw;
                float val = acc[a][v] + Bv[co];
                if (R) val += (float)R[oidx];
                if (relu_out) val = val > 0.f ? val : 0.f;
                Y[oidx] = (TO)val;
            }
        }
    }
}

// ---------------------------------------------------------------------------
// Vector quantization: per token argmin over 512 codes (D=64), LDS-staged
// codebook, dist = ||c||^2 - 2 x.c (||x||^2 constant for argmin).
// z stays fp32 (loss-sensitive); quantized output q is bf16 for the decoder.
// ---------------------------------------------------------------------------
__global__ __launch_bounds__(256) void vq_argmin_kernel(
    const float* __restrict__ Z, const float* __restrict__ CB,
    __bf16* __restrict__ Q, unsigned int* __restrict__ counts,
    float* __restrict__ sse)
{
    __shared__ float cs[64 * 64];   // 64 codes x 64 dims
    __shared__ float cc[64];        // ||c||^2

    const int t  = blockIdx.x * 256 + threadIdx.x;   // token id, HW=4096
    const int n  = t >> 12;
    const int hw = t & 4095;

    float x[64];
    #pragma unroll
    for (int d = 0; d < 64; ++d) x[d] = Z[((n << 6) + d) * 4096 + hw];

    float best = 3.4e38f;
    int   bidx = 0;
    for (int c0 = 0; c0 < 512; c0 += 64) {
        for (int i = threadIdx.x; i < 64 * 64; i += 256) cs[i] = CB[c0 * 64 + i];
        if (threadIdx.x < 64) {
            float s = 0.f;
            #pragma unroll
            for (int d = 0; d < 64; ++d) {
                float cv = CB[(c0 + threadIdx.x) * 64 + d];
                s = fmaf(cv, cv, s);
            }
            cc[threadIdx.x] = s;
        }
        __syncthreads();
        for (int c = 0; c < 64; ++c) {
            const float* cp = &cs[c * 64];   // uniform address -> LDS broadcast
            float dot = 0.f;
            #pragma unroll
            for (int d = 0; d < 64; ++d) dot = fmaf(x[d], cp[d], dot);
            float dist = cc[c] - 2.f * dot;
            if (dist < best) { best = dist; bidx = c0 + c; }
        }
        __syncthreads();
    }

    float lsse = 0.f;
    #pragma unroll
    for (int d = 0; d < 64; ++d) {
        float qd = CB[bidx * 64 + d];
        Q[((n << 6) + d) * 4096 + hw] = (__bf16)qd;
        float df = qd - x[d];
        lsse = fmaf(df, df, lsse);
    }
    atomicAdd(&counts[bidx], 1u);
    for (int off = 16; off > 0; off >>= 1)
        lsse += __shfl_down(lsse, off, 32);
    if ((threadIdx.x & 31) == 0) atomicAdd(sse, lsse);
}

__global__ void vq_finalize_kernel(const unsigned int* __restrict__ counts,
                                   const float* __restrict__ sse,
                                   float* __restrict__ loss_out,
                                   float* __restrict__ ppl_out,
                                   float invT, float invTD)
{
    __shared__ float red[512];
    const int i = threadIdx.x;
    float p = (float)counts[i] * invT;
    red[i] = p * __logf(p + 1e-10f);
    __syncthreads();
    for (int s = 256; s > 0; s >>= 1) {
        if (i < s) red[i] += red[i + s];
        __syncthreads();
    }
    if (i == 0) {
        *ppl_out  = __expf(-red[0]);
        *loss_out = sse[0] * 1.25f * invTD;   // loss_q + beta*loss_e, beta=0.25
    }
}

// ---------------------------------------------------------------------------
#define LAUNCH_CONV(TI, TO, KH, KW, ST, PD, LD, WOB, HWOB, Xp, Wp, Bp, Rp, Yp, Nn, Ci, Hh, Ww, Co, ri, ro) \
    do {                                                                                                   \
        int P_ = (Nn) << (HWOB);                                                                           \
        dim3 g_((P_ + TILE_P - 1) / TILE_P, ((Co) + 15) / 16);                                             \
        conv_wmma_bf16<TI, TO, KH, KW, ST, PD, LD, WOB, HWOB>                                              \
            <<<g_, 256, 0, stream>>>(Xp, Wp, Bp, Rp, Yp, Nn, Ci, Hh, Ww, Co, ri, ro);                      \
    } while (0)

extern "C" void kernel_launch(void* const* d_in, const int* in_sizes, int n_in,
                              void* d_out, int out_size, void* d_ws, size_t ws_size,
                              hipStream_t stream) {
    (void)in_sizes; (void)n_in; (void)out_size; (void)ws_size;

    const float* x        = (const float*)d_in[0];
    const float* enc_w1   = (const float*)d_in[1];
    const float* enc_b1   = (const float*)d_in[2];
    const float* enc_w2   = (const float*)d_in[3];
    const float* enc_b2   = (const float*)d_in[4];
    const float* enc_w3   = (const float*)d_in[5];
    const float* enc_b3   = (const float*)d_in[6];
    const float* enc_rw3  = (const float*)d_in[7];
    const float* enc_rb3  = (const float*)d_in[8];
    const float* enc_rw1  = (const float*)d_in[9];
    const float* enc_rb1  = (const float*)d_in[10];
    const float* pre_w    = (const float*)d_in[11];
    const float* pre_b    = (const float*)d_in[12];
    const float* codebook = (const float*)d_in[13];
    const float* dec_w1   = (const float*)d_in[14];
    const float* dec_b1   = (const float*)d_in[15];
    const float* dec_rw3  = (const float*)d_in[16];
    const float* dec_rb3  = (const float*)d_in[17];
    const float* dec_rw1  = (const float*)d_in[18];
    const float* dec_rb1  = (const float*)d_in[19];
    const float* dec_dw1  = (const float*)d_in[20];
    const float* dec_db1  = (const float*)d_in[21];
    const float* dec_dw2  = (const float*)d_in[22];
    const float* dec_db2  = (const float*)d_in[23];

    // ---- workspace carve-up: bf16 activations, fp32 z + accumulators ------
    unsigned char* base = (unsigned char*)d_ws;
    size_t o = 0;
    __bf16* bufA = (__bf16*)(base + o); o += (size_t)16 * 64 * 128 * 128 * 2;   // 33.5 MB
    __bf16* bufB = (__bf16*)(base + o); o += (size_t)16 * 128 * 64 * 64 * 2;    // 16.8 MB
    __bf16* bufC = (__bf16*)(base + o); o += (size_t)16 * 128 * 64 * 64 * 2;    // 16.8 MB
    __bf16* bufT = (__bf16*)(base + o); o += (size_t)16 * 32 * 64 * 64 * 2;     //  4.2 MB
    __bf16* bufQ = (__bf16*)(base + o); o += (size_t)16 * 64 * 64 * 64 * 2;     //  8.4 MB
    float*  bufZ = (float*)(base + o);  o += (size_t)16 * 64 * 64 * 64 * 4;     // 16.8 MB (fp32)
    unsigned int* counts = (unsigned int*)(base + o); o += 512 * 4;
    float* sse = (float*)(base + o); o += 4;

    hipMemsetAsync(counts, 0, 513 * sizeof(unsigned int), stream);

    float* out  = (float*)d_out;
    float* loss = out;
    float* rec  = out + 1;                       // 16*3*256*256 = 3145728
    float* ppl  = out + 1 + (size_t)16 * 3 * 256 * 256;

    const __bf16* NOR = nullptr;                 // no residual (bf16 layers)
    const float*  NORF = nullptr;                // no residual (fp32 layers)

    // ---------------- Encoder ----------------
    // h = relu(conv(x, enc_w1, s2 p1 k4))   [16,64,128,128]
    LAUNCH_CONV(float,  __bf16, 4,4,2,1,1, 7,14, x,    enc_w1, enc_b1, NOR, bufA, 16,   3, 256, 256,  64, 0, 1);
    // h = relu(conv(h, enc_w2, s2 p1 k4))   [16,128,64,64]
    LAUNCH_CONV(__bf16, __bf16, 4,4,2,1,1, 6,12, bufA, enc_w2, enc_b2, NOR, bufB, 16,  64, 128, 128, 128, 0, 1);
    // h = conv(h, enc_w3, s1 p1 k3)         [16,128,64,64]
    LAUNCH_CONV(__bf16, __bf16, 3,3,1,1,1, 6,12, bufB, enc_w3, enc_b3, NOR, bufC, 16, 128,  64,  64, 128, 0, 0);
    // residual stack x2: x += conv1x1(relu(conv3x3(relu(x))))
    for (int i = 0; i < 2; ++i) {
        LAUNCH_CONV(__bf16, __bf16, 3,3,1,1,1, 6,12, bufC, enc_rw3 + (size_t)i * 32 * 128 * 9,
                    enc_rb3 + (size_t)i * 32, NOR, bufT, 16, 128, 64, 64, 32, 1, 1);
        LAUNCH_CONV(__bf16, __bf16, 1,1,1,0,1, 6,12, bufT, enc_rw1 + (size_t)i * 128 * 32,
                    enc_rb1 + (size_t)i * 128, (const __bf16*)bufC, bufC, 16, 32, 64, 64, 128, 0, 0);
    }
    // z = conv1x1(relu(x), pre_w)  (fuses the residual_stack's final relu); fp32 out for VQ
    LAUNCH_CONV(__bf16, float, 1,1,1,0,1, 6,12, bufC, pre_w, pre_b, NORF, bufZ, 16, 128, 64, 64, 64, 1, 0);

    // ---------------- Vector quantization ----------------
    vq_argmin_kernel<<<65536 / 256, 256, 0, stream>>>(bufZ, codebook, bufQ, counts, sse);

    // ---------------- Decoder ----------------
    // d = conv(q, dec_w1, s1 p1 k3)         [16,128,64,64]
    LAUNCH_CONV(__bf16, __bf16, 3,3,1,1,1, 6,12, bufQ, dec_w1, dec_b1, NOR, bufC, 16, 64, 64, 64, 128, 0, 0);
    for (int i = 0; i < 2; ++i) {
        LAUNCH_CONV(__bf16, __bf16, 3,3,1,1,1, 6,12, bufC, dec_rw3 + (size_t)i * 32 * 128 * 9,
                    dec_rb3 + (size_t)i * 32, NOR, bufT, 16, 128, 64, 64, 32, 1, 1);
        LAUNCH_CONV(__bf16, __bf16, 1,1,1,0,1, 6,12, bufT, dec_rw1 + (size_t)i * 128 * 32,
                    dec_rb1 + (size_t)i * 128, (const __bf16*)bufC, bufC, 16, 32, 64, 64, 128, 0, 0);
    }
    // d = relu(conv_t(relu(d), dec_dw1))  = lhs-dilated conv, pad=2, k4  [16,64,128,128]
    LAUNCH_CONV(__bf16, __bf16, 4,4,1,2,2, 7,14, bufC, dec_dw1, dec_db1, NOR, bufA, 16, 128,  64,  64, 64, 1, 1);
    // x_recon = conv_t(d, dec_dw2)  -> fp32 output                      [16,3,256,256]
    LAUNCH_CONV(__bf16, float, 4,4,1,2,2, 8,16, bufA, dec_dw2, dec_db2, NORF, rec, 16,  64, 128, 128,  3, 0, 0);

    // ---------------- loss / perplexity ----------------
    vq_finalize_kernel<<<1, 512, 0, stream>>>(counts, sse, loss, ppl,
                                              1.0f / 65536.0f,
                                              1.0f / (65536.0f * 64.0f));
}
